// MHSA_Intra_3281355014316
// MI455X (gfx1250) — matrix operand order repaired and checked by reading, not compiled, not run on hardware
//
#include <hip/hip_runtime.h>
#include <hip/hip_bf16.h>
#include <math.h>

#define B_ 4
#define C_ 512
#define T_ 2048
#define H_ 8
#define D_ 64

typedef __attribute__((ext_vector_type(16))) _Float16 v16h;
typedef __attribute__((ext_vector_type(8)))  float    v8f;

__device__ __forceinline__ v8f wmma_f16(v16h a, v16h b, v8f c) {
  // D = A(16x32 f16) * B(32x16 f16) + C(16x16 f32)
  return __builtin_amdgcn_wmma_f32_16x16x32_f16(false, a, false, b, (short)0, c,
                                                false, false);
}

// Raw v_exp_f32 (base-2 exponential)
__device__ __forceinline__ float fast_exp2(float x) {
#if __has_builtin(__builtin_amdgcn_exp2f)
  return __builtin_amdgcn_exp2f(x);
#else
  return exp2f(x);
#endif
}

// Async global -> LDS copy (16B per lane), tracked by ASYNCcnt.
__device__ __forceinline__ void async_ld_b128(unsigned lds_off, const void* g) {
  asm volatile("global_load_async_to_lds_b128 %0, %1, off"
               :: "v"(lds_off), "v"(g) : "memory");
}

// ---- cross-lane max over the 16-lane N group (VALU dpp/permlane, no LDS) --
constexpr unsigned dpp8sel(int a, int b, int c, int d, int e, int f, int g, int h) {
  return (unsigned)a | ((unsigned)b << 3) | ((unsigned)c << 6) | ((unsigned)d << 9) |
         ((unsigned)e << 12) | ((unsigned)f << 15) | ((unsigned)g << 18) |
         ((unsigned)h << 21);
}

__device__ __forceinline__ float redmax16(float x) {
#if __has_builtin(__builtin_amdgcn_mov_dpp8)
  x = fmaxf(x, __int_as_float(__builtin_amdgcn_mov_dpp8(
                   __float_as_int(x), dpp8sel(1, 0, 3, 2, 5, 4, 7, 6))));  // xor1
  x = fmaxf(x, __int_as_float(__builtin_amdgcn_mov_dpp8(
                   __float_as_int(x), dpp8sel(2, 3, 0, 1, 6, 7, 4, 5))));  // xor2
  x = fmaxf(x, __int_as_float(__builtin_amdgcn_mov_dpp8(
                   __float_as_int(x), dpp8sel(4, 5, 6, 7, 0, 1, 2, 3))));  // xor4
#else
  x = fmaxf(x, __shfl_xor(x, 1, 32));
  x = fmaxf(x, __shfl_xor(x, 2, 32));
  x = fmaxf(x, __shfl_xor(x, 4, 32));
#endif
#if __has_builtin(__builtin_amdgcn_permlane16)
  x = fmaxf(x, __int_as_float(__builtin_amdgcn_permlane16(
                   __float_as_int(x), __float_as_int(x), 0xFEDCBA98u, 0x76543210u,
                   false, false)));                                        // xor8
#else
  x = fmaxf(x, __shfl_xor(x, 8, 32));
#endif
  return x;
}

// ---- Fragment loaders (contiguous k = 16*hi + i, same order in A and B) ----------

// A fragment: 16(M) x 32(K) f16, source row-major [M][lda].
__device__ __forceinline__ v16h load_a_rm(const _Float16* p, int lda, int lane) {
  const int l = lane & 15, hi = lane >> 4;
  const _Float16* row = p + (size_t)l * lda;
  v16h a;
#pragma unroll
  for (int i = 0; i < 8; ++i) a[i] = row[hi * 16 + i];
#pragma unroll
  for (int i = 0; i < 8; ++i) a[8 + i] = row[hi * 16 + 8 + i];
  return a;
}

// B fragment: 32(K) x 16(N) f16, source row-major [K][ldk] (K-major)
__device__ __forceinline__ v16h load_b_km(const _Float16* p, int ldk, int lane) {
  const int l = lane & 15, hi = lane >> 4;
  const _Float16* base = p + (size_t)(hi * 16) * ldk + l;
  v16h b;
#pragma unroll
  for (int i = 0; i < 16; ++i) b[i] = base[(size_t)i * ldk];
  return b;
}

// B fragment where source is stored N-major (B^T row-major): elem(k,n)=p[n*ldn+k]
// -> 32 contiguous bytes per lane (lowered to wide global loads)
__device__ __forceinline__ v16h load_b_nm(const _Float16* p, int ldn, int lane) {
  const int l = lane & 15, hi = lane >> 4;
  const _Float16* col = p + (size_t)l * ldn + hi * 16;
  v16h b;
#pragma unroll
  for (int i = 0; i < 16; ++i) b[i] = col[i];
  return b;
}

// ---- f32 -> f16 conversion ------------------------------------------------
__global__ __launch_bounds__(256)
void cvt_f16(const float* __restrict__ src, _Float16* __restrict__ dst, int n) {
  const int i = blockIdx.x * 256 + threadIdx.x;
  if (i < n) dst[i] = (_Float16)src[i];
}

// ---- Projection GEMM: Y[o,t] = sum_c W[o,c]*X[b,c,t] + bias[o] ------------
// MODE 0: f32 out [B,C,T];  MODE 1: f16 out [B,H,T,D];  MODE 2: f16 out [B,C,T]
// grid (T/64, C/128, B), 256 threads = 8 waves, wave -> 16(M)x64(N) strip.
// X tiles double-buffered through LDS via async global->LDS (ASYNCcnt).
template<int MODE>
__global__ __launch_bounds__(256)
void proj_gemm(const _Float16* __restrict__ Wh,   // [C,C] row-major (o x c)
               const _Float16* __restrict__ Xh,   // [B,C,T]
               const float*    __restrict__ bias, // [C]
               _Float16*       __restrict__ outh,
               float*          __restrict__ outf)
{
  __shared__ _Float16 xs[2][32 * 72];              // 32(K) x 64(N), stride 72
  const int t0   = blockIdx.x * 64;
  const int m0   = blockIdx.y * 128 + (threadIdx.x >> 5) * 16;
  const int b    = blockIdx.z;
  const int lane = threadIdx.x & 31;
  const int l = lane & 15, hi = lane >> 4;
  const int row = threadIdx.x >> 3;                // K row of staged tile
  const int ch  = threadIdx.x & 7;                 // 8-half chunk along N

  v8f acc[4] = {v8f{}, v8f{}, v8f{}, v8f{}};

  const _Float16* gsrc = Xh + ((size_t)(b * C_ + row) * T_ + t0 + ch * 8);
  const unsigned ldsoff[2] = {
      (unsigned)(uintptr_t)&xs[0][row * 72 + ch * 8],
      (unsigned)(uintptr_t)&xs[1][row * 72 + ch * 8]};

  async_ld_b128(ldsoff[0], gsrc);                  // prologue: tile k0=0
  int buf = 0;
  for (int k0 = 0; k0 < C_; k0 += 32, buf ^= 1) {
    if (k0 + 32 < C_) {
      async_ld_b128(ldsoff[buf ^ 1], gsrc + (size_t)(k0 + 32) * T_);
      asm volatile("s_wait_asynccnt 0x1" ::: "memory");   // current tile done
    } else {
      asm volatile("s_wait_asynccnt 0x0" ::: "memory");
    }
    __syncthreads();

    const v16h a = load_a_rm(Wh + (size_t)m0 * C_ + k0, C_, lane);
#pragma unroll
    for (int j = 0; j < 4; ++j)
      acc[j] = wmma_f16(a, load_b_km(&xs[buf][j * 16], 72, lane), acc[j]);
    __syncthreads();   // all waves done with xs[buf] before it is re-filled
  }

#pragma unroll
  for (int r = 0; r < 8; ++r) {
    const int o = m0 + r + 8 * hi;
    const float bi = bias[o];
#pragma unroll
    for (int j = 0; j < 4; ++j) {
      const int t = t0 + j * 16 + l;
      const float yv = acc[j][r] + bi;
      if constexpr (MODE == 1) {
        const int h = o >> 6, d = o & 63;
        outh[(((size_t)b * H_ + h) * T_ + t) * D_ + d] = (_Float16)yv;
      } else if constexpr (MODE == 2) {
        outh[((size_t)b * C_ + o) * T_ + t] = (_Float16)yv;
      } else {
        outf[((size_t)b * C_ + o) * T_ + t] = yv;
      }
    }
  }
}

// ---- Flash attention: one wave per (b,h,16-query-row tile) ----------------
// q,k: [B,H,T,D] f16.  v: [B,C,T] f16 (d-major!).  ao: [B,C,T] f16.
// All logits kept in log2 domain: Q pre-scaled by D^-1/2 * log2(e), exp == exp2.
__global__ __launch_bounds__(256)
void attn_kernel(const _Float16* __restrict__ q,
                 const _Float16* __restrict__ k,
                 const _Float16* __restrict__ v,
                 const int*      __restrict__ mask,  // [B,T,T]
                 _Float16*       __restrict__ ao)
{
  __shared__ _Float16 ps[8][16][40];   // per-wave P tile, padded stride
  const int wave = threadIdx.x >> 5;
  const int lane = threadIdx.x & 31;
  const int l = lane & 15, hi = lane >> 4;

  const int w   = blockIdx.x * 8 + wave;
  const int qt  = w & 127;
  const int bh  = w >> 7;
  const int h   = bh & 7, b = bh >> 3;
  const int tq0 = qt * 16;

  const _Float16* qp = q + (((size_t)b * H_ + h) * T_ + tq0) * D_;
  const _Float16* kp = k + (((size_t)b * H_ + h) * T_) * D_;
  const _Float16* vp = v + ((size_t)b * C_ + h * 64) * T_;  // [64][T] d-major
  const int*      mp = mask + (size_t)b * T_ * T_;

  v16h qa0 = load_a_rm(qp,      D_, lane);
  v16h qa1 = load_a_rm(qp + 32, D_, lane);
  const _Float16 sc = (_Float16)(0.125f * 1.44269504f);  // D^-1/2 * log2(e)
#pragma unroll
  for (int i = 0; i < 16; ++i) { qa0[i] *= sc; qa1[i] *= sc; }

  v16h ones;
#pragma unroll
  for (int i = 0; i < 16; ++i) ones[i] = (_Float16)1.0f;

  v8f o0 = {}, o1 = {}, o2 = {}, o3 = {};
  const float NEG_BIG = -1e30f;          // finite sentinel: no NaN guards needed
  float mrow[8], lrow[8];
#pragma unroll
  for (int r = 0; r < 8; ++r) { mrow[r] = NEG_BIG; lrow[r] = 0.f; }

  for (int kv0 = 0; kv0 < T_; kv0 += 32) {
    // S = Q * K^T for 32 keys (two 16x16 tiles, K-dim 64 = 2 x k32)
    v8f s0 = {}, s1 = {};
    s0 = wmma_f16(qa0, load_b_nm(kp + (size_t)kv0 * D_,           D_, lane), s0);
    s0 = wmma_f16(qa1, load_b_nm(kp + (size_t)kv0 * D_ + 32,      D_, lane), s0);
    s1 = wmma_f16(qa0, load_b_nm(kp + (size_t)(kv0 + 16) * D_,      D_, lane), s1);
    s1 = wmma_f16(qa1, load_b_nm(kp + (size_t)(kv0 + 16) * D_ + 32, D_, lane), s1);

    // masked online softmax (base-2); C layout: VGPR r <-> row r+8*hi, col = l
    float alpha[8];
#pragma unroll
    for (int r = 0; r < 8; ++r) {
      const int tq  = tq0 + r + 8 * hi;
      const size_t mrowoff = (size_t)tq * T_ + kv0;
      const int mk0 = mp[mrowoff + l];
      const int mk1 = mp[mrowoff + 16 + l];
      __builtin_prefetch(mp + mrowoff + 32 + l, 0, 3);   // next kv step, near$
      const float s0v = mk0 ? s0[r] : NEG_BIG;
      const float s1v = mk1 ? s1[r] : NEG_BIG;
      const float mx = redmax16(fmaxf(s0v, s1v));        // dpp8+permlane, no LDS
      const float mn = fmaxf(mrow[r], mx);
      alpha[r] = fast_exp2(mrow[r] - mn);                // ==1 when nothing seen
      const float p0 = mk0 ? fast_exp2(s0v - mn) : 0.f;
      const float p1 = mk1 ? fast_exp2(s1v - mn) : 0.f;
      mrow[r] = mn;
      ps[wave][r + 8 * hi][l]      = (_Float16)p0;
      ps[wave][r + 8 * hi][l + 16] = (_Float16)p1;
    }
    // same-wave LDS ops are in-order (DScnt) -> no barrier needed
    const v16h pa = load_a_rm(&ps[wave][0][0], 40, lane);

    // row sums via WMMA: D[r] = sum_k P[row][k]  (lands in the lrow[r] slot)
    const v8f zero = {};
    const v8f sums = wmma_f16(pa, ones, zero);
#pragma unroll
    for (int r = 0; r < 8; ++r) {
      lrow[r] = lrow[r] * alpha[r] + sums[r];
      o0[r] *= alpha[r]; o1[r] *= alpha[r]; o2[r] *= alpha[r]; o3[r] *= alpha[r];
    }

    // O += P(16x32) * V(32x64); V is d-major -> contiguous 32B per lane
    o0 = wmma_f16(pa, load_b_nm(vp + (size_t) 0 * T_ + kv0, T_, lane), o0);
    o1 = wmma_f16(pa, load_b_nm(vp + (size_t)16 * T_ + kv0, T_, lane), o1);
    o2 = wmma_f16(pa, load_b_nm(vp + (size_t)32 * T_ + kv0, T_, lane), o2);
    o3 = wmma_f16(pa, load_b_nm(vp + (size_t)48 * T_ + kv0, T_, lane), o3);
  }

  float inv[8];
#pragma unroll
  for (int r = 0; r < 8; ++r) inv[r] = (lrow[r] > 0.f) ? (1.f / lrow[r]) : 0.f;

#pragma unroll
  for (int r = 0; r < 8; ++r) {
    const int t = tq0 + r + 8 * hi;
    const size_t cb = (size_t)b * C_ + h * 64 + l;
    ao[(cb +  0) * T_ + t] = (_Float16)(o0[r] * inv[r]);
    ao[(cb + 16) * T_ + t] = (_Float16)(o1[r] * inv[r]);
    ao[(cb + 32) * T_ + t] = (_Float16)(o2[r] * inv[r]);
    ao[(cb + 48) * T_ + t] = (_Float16)(o3[r] * inv[r]);
  }
}

// ---- BatchNorm stats: per-channel mean & rsqrt(var+eps) over (B,T) --------
__global__ __launch_bounds__(256)
void bn_stats(const float* __restrict__ y, float* __restrict__ mean,
              float* __restrict__ rsig) {
  __shared__ float sh1[256], sh2[256];
  const int c = blockIdx.x, tid = threadIdx.x;
  float s1 = 0.f, s2 = 0.f;
  for (int b = 0; b < B_; ++b) {
    const float* p = y + ((size_t)b * C_ + c) * T_;
    for (int t = tid; t < T_; t += 256) { const float vv = p[t]; s1 += vv; s2 += vv * vv; }
  }
  sh1[tid] = s1; sh2[tid] = s2;
  __syncthreads();
  for (int s = 128; s > 0; s >>= 1) {
    if (tid < s) { sh1[tid] += sh1[tid + s]; sh2[tid] += sh2[tid + s]; }
    __syncthreads();
  }
  if (tid == 0) {
    const float n = (float)(B_ * T_);
    const float m = sh1[0] / n;
    const float var = sh2[0] / n - m * m;     // biased, matches reference
    mean[c] = m;
    rsig[c] = rsqrtf(var + 1e-5f);
  }
}

// ---- out = input + gamma*(y-mean)*rsig + beta -----------------------------
__global__ __launch_bounds__(256)
void bn_apply(const float* __restrict__ x, const float* __restrict__ y,
              const float* __restrict__ mean, const float* __restrict__ rsig,
              const float* __restrict__ gamma, const float* __restrict__ beta,
              float* __restrict__ out) {
  const int i = blockIdx.x * 256 + threadIdx.x;
  const int c = (i >> 11) & (C_ - 1);          // T = 2^11
  out[i] = x[i] + gamma[c] * (y[i] - mean[c]) * rsig[c] + beta[c];
}

// ---- host side ------------------------------------------------------------
extern "C" void kernel_launch(void* const* d_in, const int* in_sizes, int n_in,
                              void* d_out, int out_size, void* d_ws, size_t ws_size,
                              hipStream_t stream) {
  (void)in_sizes; (void)n_in; (void)out_size; (void)ws_size;
  const float* x     = (const float*)d_in[0];
  const int*   mask  = (const int*)  d_in[1];
  const float* Wq    = (const float*)d_in[2];
  const float* bq    = (const float*)d_in[3];
  const float* Wk    = (const float*)d_in[4];
  const float* bk    = (const float*)d_in[5];
  const float* Wv    = (const float*)d_in[6];
  const float* bv    = (const float*)d_in[7];
  const float* Wo    = (const float*)d_in[8];
  const float* bo    = (const float*)d_in[9];
  const float* gamma = (const float*)d_in[10];
  const float* beta  = (const float*)d_in[11];
  float* out = (float*)d_out;

  constexpr size_t NCT = (size_t)B_ * C_ * T_;   // 4,194,304
  constexpr size_t NW  = (size_t)C_ * C_;        // 262,144

  char* ws = (char*)d_ws;
  size_t off = 0;
  auto carve = [&](size_t bytes) { char* p = ws + off; off = (off + bytes + 255) & ~(size_t)255; return p; };
  _Float16* xh   = (_Float16*)carve(NCT * 2);
  _Float16* wqh  = (_Float16*)carve(NW * 2);
  _Float16* wkh  = (_Float16*)carve(NW * 2);
  _Float16* wvh  = (_Float16*)carve(NW * 2);
  _Float16* woh  = (_Float16*)carve(NW * 2);
  _Float16* qb   = (_Float16*)carve(NCT * 2);
  _Float16* kb   = (_Float16*)carve(NCT * 2);
  _Float16* vb   = (_Float16*)carve(NCT * 2);
  _Float16* aob  = (_Float16*)carve(NCT * 2);
  float*    yb   = (float*)   carve(NCT * 4);
  float*    mnb  = (float*)   carve(C_ * 4);
  float*    rsb  = (float*)   carve(C_ * 4);

  const dim3 blk(256);
  cvt_f16<<<(int)(NCT / 256), blk, 0, stream>>>(x,  xh,  (int)NCT);
  cvt_f16<<<(int)(NW  / 256), blk, 0, stream>>>(Wq, wqh, (int)NW);
  cvt_f16<<<(int)(NW  / 256), blk, 0, stream>>>(Wk, wkh, (int)NW);
  cvt_f16<<<(int)(NW  / 256), blk, 0, stream>>>(Wv, wvh, (int)NW);
  cvt_f16<<<(int)(NW  / 256), blk, 0, stream>>>(Wo, woh, (int)NW);

  const dim3 ggrid(T_ / 64, C_ / 128, B_);
  proj_gemm<1><<<ggrid, blk, 0, stream>>>(wqh, xh, bq, qb, nullptr);   // Q [B,H,T,D]
  proj_gemm<1><<<ggrid, blk, 0, stream>>>(wkh, xh, bk, kb, nullptr);   // K [B,H,T,D]
  proj_gemm<2><<<ggrid, blk, 0, stream>>>(wvh, xh, bv, vb, nullptr);   // V [B,C,T] d-major

  attn_kernel<<<(B_ * H_ * (T_ / 16)) / 8, blk, 0, stream>>>(qb, kb, vb, mask, aob);

  proj_gemm<0><<<ggrid, blk, 0, stream>>>(woh, aob, bo, nullptr, yb);

  bn_stats<<<C_, blk, 0, stream>>>(yb, mnb, rsb);
  bn_apply<<<(int)(NCT / 256), blk, 0, stream>>>(x, yb, mnb, rsb, gamma, beta, out);
}
